// E62SelectiveWriteCell_4887672783137
// MI455X (gfx1250) — matrix operand order repaired and checked by reading, not compiled, not run on hardware
//
#include <hip/hip_runtime.h>

// ---------------------------------------------------------------------------
// E62SelectiveWriteCell for MI455X (gfx1250, wave32, WMMA, LDS double-buffer)
//
// k = sigmoid(x @ Wk^T + bk); v = tanh(x @ Wv^T + bv)   -> bf16 WMMA GEMMs
// scan: h = (1-k)h + k v  (sequential over T, parallel over B*D)
// out  = h * silu(h)
//
// d_out = [ output (T*B*D f32) | h ((T+1)*B*D f32) ]
// GEMM writes k_all into the output region and v_all into h rows 1..T; the
// scan consumes each element before overwriting it in place, so the only
// workspace needed is 68MB of bf16 (x, Wk, Wv).
// ---------------------------------------------------------------------------

typedef __attribute__((ext_vector_type(16))) __bf16 v16bf;
typedef __attribute__((ext_vector_type(8)))  __bf16 v8bf;
typedef __attribute__((ext_vector_type(4)))  __bf16 v4bf;
typedef __attribute__((ext_vector_type(8)))  float  v8f;

#define T_DIM 2048
#define B_DIM 16
#define D_DIM 1024
#define M_DIM (T_DIM * B_DIM)      /* 32768 rows of the batched GEMM  */
#define BD    (B_DIM * D_DIM)      /* 16384 independent scan columns  */
#define NW    (D_DIM * D_DIM)      /* 1048576 weight elements         */

#define BLK_M 256                  /* block tile rows                  */
#define BLK_N 64                   /* block tile cols (per gate)       */
#define BK    32                   /* K step = WMMA K                  */
#define LDA   40                   /* padded LDS row stride (elements) */
#define NCHUNK (D_DIM / BK)        /* 32 K-chunks                      */

static __device__ __forceinline__ v16bf cat16(v8bf lo, v8bf hi) {
  return __builtin_shufflevector(lo, hi, 0, 1, 2, 3, 4, 5, 6, 7,
                                         8, 9, 10, 11, 12, 13, 14, 15);
}

// ---------------------------------------------------------------------------
// Kernel 1: fp32 -> bf16 conversion of Wk | Wv | x into workspace (vec4).
// dst layout: [ Wk bf16 (NW) | Wv bf16 (NW) | x bf16 (M*D) ]
// ---------------------------------------------------------------------------
__global__ __launch_bounds__(256) void cvt_bf16_kernel(
    const float* __restrict__ wk, const float* __restrict__ wv,
    const float* __restrict__ x, unsigned short* __restrict__ dst_raw) {
  const size_t total = 2ull * NW + (size_t)M_DIM * D_DIM;
  size_t e = ((size_t)blockIdx.x * blockDim.x + threadIdx.x) * 4ull;
  if (e >= total) return;

  const float* src;
  if (e < (size_t)NW)            src = wk + e;
  else if (e < 2ull * NW)        src = wv + (e - NW);
  else                           src = x  + (e - 2ull * NW);

  float4 f = *(const float4*)src;
  v4bf o;
  o[0] = (__bf16)f.x;
  o[1] = (__bf16)f.y;
  o[2] = (__bf16)f.z;
  o[3] = (__bf16)f.w;
  *(v4bf*)(dst_raw + e) = o;
}

// ---------------------------------------------------------------------------
// Kernel 2: dual-gate bf16 WMMA GEMM, LDS double-buffered.
// 256 threads = 8 waves. Block tile 256(M) x 64(N) x 2 gates.
// Wave w owns rows m_base + w*32 (TWO 16-row m-tiles): each B fragment read
// from LDS feeds two WMMAs, halving LDS bytes per matrix op. 16 WMMA per
// wave per 32-wide K chunk; next chunk's global loads are in flight during
// compute and land in the other LDS buffer.
//
// Fragment layouts per CDNA5 ISA 7.12.2 (wave32), lane = half*16 + r:
//   A 16x32 bf16: row M=r; V0..3 = K[half*8..+7], V4..7 = K[16+half*8..+7]
//   B 32x16 bf16: col N=r; V0..3 = K[half*8..+7], V4..7 = K[16+half*8..+7] (same as A)
//   C/D 16x16 f32: VGPR i = (M = i + 8*half, N = r)
// ---------------------------------------------------------------------------
__global__ __launch_bounds__(256) void gate_gemm_kernel(
    const unsigned short* __restrict__ xb_raw,
    const unsigned short* __restrict__ wkb_raw,
    const unsigned short* __restrict__ wvb_raw,
    const float* __restrict__ bk, const float* __restrict__ bv,
    float* __restrict__ kDst /* output region */,
    float* __restrict__ hDst /* h region      */) {
  const __bf16* xb  = (const __bf16*)xb_raw;
  const __bf16* wkb = (const __bf16*)wkb_raw;
  const __bf16* wvb = (const __bf16*)wvb_raw;

  __shared__ __bf16 As [2][BLK_M * LDA];   // 2 x 20480 B
  __shared__ __bf16 Bks[2][BLK_N * LDA];   // 2 x  5120 B
  __shared__ __bf16 Bvs[2][BLK_N * LDA];   // 2 x  5120 B

  const int tid  = threadIdx.x;
  const int lane = tid & 31;
  const int wave = tid >> 5;
  const int rl   = lane & 15;
  const int half = lane >> 4;

  const int m_base = blockIdx.x * BLK_M;
  const int n_base = blockIdx.y * BLK_N;

  // --- cooperative loader mapping: 256 threads cover 64 rows x 4 16B segs
  const int lrow = tid >> 2;          // 0..63
  const int lseg = (tid & 3) * 8;     // element offset within K chunk

  const __bf16* aG0 = xb  + (size_t)(m_base + lrow) * D_DIM + lseg;
  const __bf16* aG1 = aG0 + (size_t)64  * D_DIM;   // rows  64..127
  const __bf16* aG2 = aG0 + (size_t)128 * D_DIM;   // rows 128..191
  const __bf16* aG3 = aG0 + (size_t)192 * D_DIM;   // rows 192..255
  const __bf16* bkG = wkb + (size_t)(n_base + lrow) * D_DIM + lseg;
  const __bf16* bvG = wvb + (size_t)(n_base + lrow) * D_DIM + lseg;

  const int sOff = lrow * LDA + lseg;

  // --- prologue: stage chunk 0
  {
    v8bf a0 = *(const v8bf*)aG0;
    v8bf a1 = *(const v8bf*)aG1;
    v8bf a2 = *(const v8bf*)aG2;
    v8bf a3 = *(const v8bf*)aG3;
    v8bf b0 = *(const v8bf*)bkG;
    v8bf b1 = *(const v8bf*)bvG;
    *(v8bf*)&As [0][sOff]             = a0;
    *(v8bf*)&As [0][64  * LDA + sOff] = a1;
    *(v8bf*)&As [0][128 * LDA + sOff] = a2;
    *(v8bf*)&As [0][192 * LDA + sOff] = a3;
    *(v8bf*)&Bks[0][sOff]             = b0;
    *(v8bf*)&Bvs[0][sOff]             = b1;
  }
  __syncthreads();

  v8f accK[2][4] = {};
  v8f accV[2][4] = {};

  const int aFragOff = (wave * 32 + rl) * LDA + half * 8;  // m-tile 0
  const int bFragOff = rl * LDA + half * 8;

#pragma unroll 2
  for (int c = 0; c < NCHUNK; ++c) {
    const int cur = c & 1;
    const int nxt = cur ^ 1;
    const bool more = (c + 1 < NCHUNK);

    // issue next chunk's global loads early (consumed only at ds_store)
    v8bf pa0, pa1, pa2, pa3, pb0, pb1;
    if (more) {
      const int k = (c + 1) * BK;
      pa0 = *(const v8bf*)(aG0 + k);
      pa1 = *(const v8bf*)(aG1 + k);
      pa2 = *(const v8bf*)(aG2 + k);
      pa3 = *(const v8bf*)(aG3 + k);
      pb0 = *(const v8bf*)(bkG + k);
      pb1 = *(const v8bf*)(bvG + k);
    }

    // A fragments for both m-tiles of this wave
    v16bf a0 = cat16(*(const v8bf*)&As[cur][aFragOff],
                     *(const v8bf*)&As[cur][aFragOff + 16]);
    v16bf a1 = cat16(*(const v8bf*)&As[cur][aFragOff + 16 * LDA],
                     *(const v8bf*)&As[cur][aFragOff + 16 * LDA + 16]);

#pragma unroll
    for (int nt = 0; nt < 4; ++nt) {
      const int bo = nt * 16 * LDA + bFragOff;
      v16bf bkf = cat16(*(const v8bf*)&Bks[cur][bo],
                        *(const v8bf*)&Bks[cur][bo + 16]);
      accK[0][nt] = __builtin_amdgcn_wmma_f32_16x16x32_bf16(
          false, a0, false, bkf, (short)0, accK[0][nt], false, false);
      accK[1][nt] = __builtin_amdgcn_wmma_f32_16x16x32_bf16(
          false, a1, false, bkf, (short)0, accK[1][nt], false, false);
      v16bf bvf = cat16(*(const v8bf*)&Bvs[cur][bo],
                        *(const v8bf*)&Bvs[cur][bo + 16]);
      accV[0][nt] = __builtin_amdgcn_wmma_f32_16x16x32_bf16(
          false, a0, false, bvf, (short)0, accV[0][nt], false, false);
      accV[1][nt] = __builtin_amdgcn_wmma_f32_16x16x32_bf16(
          false, a1, false, bvf, (short)0, accV[1][nt], false, false);
    }

    // stage next chunk into the other buffer
    if (more) {
      *(v8bf*)&As [nxt][sOff]             = pa0;
      *(v8bf*)&As [nxt][64  * LDA + sOff] = pa1;
      *(v8bf*)&As [nxt][128 * LDA + sOff] = pa2;
      *(v8bf*)&As [nxt][192 * LDA + sOff] = pa3;
      *(v8bf*)&Bks[nxt][sOff]             = pb0;
      *(v8bf*)&Bvs[nxt][sOff]             = pb1;
    }
    __syncthreads();
  }

  // --- epilogue: bias + activations, scatter to d_out staging regions
  const int m0 = m_base + wave * 32;
#pragma unroll
  for (int mt = 0; mt < 2; ++mt) {
#pragma unroll
    for (int nt = 0; nt < 4; ++nt) {
      const int col = n_base + nt * 16 + rl;
      const float bbk = bk[col];
      const float bbv = bv[col];
#pragma unroll
      for (int i = 0; i < 8; ++i) {
        const int row = m0 + mt * 16 + i + 8 * half;   // row m = t*B + b
        float gk = accK[mt][nt][i] + bbk;
        gk = 1.0f / (1.0f + __expf(-gk));              // sigmoid gate
        kDst[(size_t)row * D_DIM + col] = gk;
        float gv = tanhf(accV[mt][nt][i] + bbv);       // value
        hDst[(size_t)(row + B_DIM) * D_DIM + col] = gv; // h row t+1
      }
    }
  }
}

// ---------------------------------------------------------------------------
// Kernel 3: sequential scan over T, one thread per (b,d) column.
// Reads k from the output region and v from h[t+1], overwrites both in place.
// Unrolled so future timesteps' loads issue behind the recurrence chain.
// ---------------------------------------------------------------------------
__global__ __launch_bounds__(256) void scan_kernel(
    const float* __restrict__ h0,
    float* __restrict__ outR, float* __restrict__ hR) {
  const int j = blockIdx.x * blockDim.x + threadIdx.x;  // j = b*D + d
  if (j >= BD) return;

  float h = h0[j];
  hR[j] = h;  // h[0] row
#pragma unroll 4
  for (int t = 0; t < T_DIM; ++t) {
    const size_t idx = (size_t)t * BD + j;
    const float kk = outR[idx];      // sigmoid gate (staged)
    const float vv = hR[idx + BD];   // tanh value   (staged at h[t+1])
    h = (1.0f - kk) * h + kk * vv;
    hR[idx + BD] = h;                // final h[t+1]
    const float s = 1.0f / (1.0f + __expf(-h));
    outR[idx] = h * h * s;           // h * silu(h)
  }
}

// ---------------------------------------------------------------------------
extern "C" void kernel_launch(void* const* d_in, const int* in_sizes, int n_in,
                              void* d_out, int out_size, void* d_ws, size_t ws_size,
                              hipStream_t stream) {
  (void)in_sizes; (void)n_in; (void)out_size; (void)ws_size;
  const float* x  = (const float*)d_in[0];
  const float* h0 = (const float*)d_in[1];
  const float* Wk = (const float*)d_in[2];
  const float* bk = (const float*)d_in[3];
  const float* Wv = (const float*)d_in[4];
  const float* bv = (const float*)d_in[5];

  float* outR = (float*)d_out;                       // [T*B*D]
  float* hR   = outR + (size_t)M_DIM * D_DIM;        // [(T+1)*B*D]

  unsigned short* wkb = (unsigned short*)d_ws;       // bf16 Wk
  unsigned short* wvb = wkb + (size_t)NW;            // bf16 Wv
  unsigned short* xb  = wvb + (size_t)NW;            // bf16 x

  // 1) fp32 -> bf16 staging (68 MB of ws)
  const size_t total  = 2ull * NW + (size_t)M_DIM * D_DIM;
  const size_t blocks = (total / 4 + 255) / 256;
  cvt_bf16_kernel<<<(unsigned)blocks, 256, 0, stream>>>(Wk, Wv, x, wkb);

  // 2) dual-gate bf16 WMMA GEMM: 128 M-tiles x 16 N-tiles, 256 thr/block
  dim3 grid(M_DIM / BLK_M, D_DIM / BLK_N);
  gate_gemm_kernel<<<grid, 256, 0, stream>>>(xb, wkb, wvb, bk, bv, outR, hR);

  // 3) sequential scan + silu epilogue
  scan_kernel<<<BD / 256, 256, 0, stream>>>(h0, outR, hR);
}
